// TokenChoiceTopKRouter_35751307772686
// MI455X (gfx1250) — matrix operand order repaired and checked by reading, not compiled, not run on hardware
//
#include <hip/hip_runtime.h>
#include <hip/hip_bf16.h>

// ---------------------------------------------------------------------------
// MoE Token-Choice Top-K router, fused for MI455X (gfx1250, wave32, WMMA).
//   logits = x @ W^T   (N=16384 tokens, D=4096, E=64 experts)
//   softmax(fp32) -> top-2 weights, stable expert-grouping permutation,
//   per-expert histogram.
// GEMM: v_wmma_f32_16x16x32_bf16.  W pre-converted to bf16 once (512 KB,
// L2-resident), then each 4 KB k-slice is staged into a double-buffered LDS
// slot via async global->LDS loads (ASYNCcnt) overlapped with WMMA compute.
// ---------------------------------------------------------------------------

typedef __attribute__((ext_vector_type(16))) __bf16 v16bf;
typedef __attribute__((ext_vector_type(8)))  float  v8f;
typedef __attribute__((ext_vector_type(4)))  int    v4i;

#define ROUTER_DIM    4096
#define ROUTER_NEXP   64
#define TPB_TOKENS    128          // tokens per block (8 waves x 16 tokens)
#define BLOCK_THREADS 256

#define HAS_ASYNC_LDS  __has_builtin(__builtin_amdgcn_global_load_async_to_lds_b128)
#define HAS_WAIT_ASYNC __has_builtin(__builtin_amdgcn_s_wait_asynccnt)

// stage 4 KB slice W_bf16[:, kb:kb+32] -> LDS [64][32] bf16; one b128/thread
__device__ __forceinline__
void stage_w_slice(const __bf16* __restrict__ Wbf, __bf16* ldsDst, int kb, int tid)
{
    const int e = tid >> 2;        // expert row 0..63
    const int c = tid & 3;         // 16-byte chunk 0..3 within the 64 B row
    const __bf16* gsrc = Wbf + (size_t)e * ROUTER_DIM + kb + c * 8;
    __bf16*       ldst = ldsDst + e * 32 + c * 8;
#if HAS_ASYNC_LDS
    __builtin_amdgcn_global_load_async_to_lds_b128(
        (__attribute__((address_space(1))) v4i*)gsrc,
        (__attribute__((address_space(3))) v4i*)ldst,
        0, 0);
#else
    *reinterpret_cast<uint4*>(ldst) = *reinterpret_cast<const uint4*>(gsrc);
#endif
}

__device__ __forceinline__ void wait_async_lds()
{
#if HAS_ASYNC_LDS
#if HAS_WAIT_ASYNC
    __builtin_amdgcn_s_wait_asynccnt(0);
#else
    asm volatile("s_wait_asynccnt 0x0" ::: "memory");
#endif
#endif
}

// ---------------------------------------------------------------------------
// one-time fp32 -> bf16 conversion of W into workspace (64 x 4096)
__global__ __launch_bounds__(BLOCK_THREADS)
void router_w_to_bf16_kernel(const float* __restrict__ W,
                             __bf16* __restrict__ Wbf, int n)
{
    const int i = (blockIdx.x * BLOCK_THREADS + threadIdx.x) * 8;
    if (i < n) {
        const float4 f0 = *reinterpret_cast<const float4*>(W + i);
        const float4 f1 = *reinterpret_cast<const float4*>(W + i + 4);
        union { __bf16 h[8]; uint4 u; } pk;
        pk.h[0] = (__bf16)f0.x; pk.h[1] = (__bf16)f0.y;
        pk.h[2] = (__bf16)f0.z; pk.h[3] = (__bf16)f0.w;
        pk.h[4] = (__bf16)f1.x; pk.h[5] = (__bf16)f1.y;
        pk.h[6] = (__bf16)f1.z; pk.h[7] = (__bf16)f1.w;
        *reinterpret_cast<uint4*>(Wbf + i) = pk.u;
    }
}

// ---------------------------------------------------------------------------
__global__ __launch_bounds__(BLOCK_THREADS)
void router_fused_kernel(const float* __restrict__ x,
                         const __bf16* __restrict__ Wbf,
                         float* __restrict__ topw_out,
                         int*   __restrict__ sel_flat,
                         int*   __restrict__ counts)
{
    // double-buffered bf16 k-slice of W: 2 x (64 experts x 32 k) = 8 KB
    __shared__ alignas(32) __bf16 ldsW[2][ROUTER_NEXP * 32];
    // logits tile: 128 tokens x 64 experts (+1 pad) ~33 KB
    __shared__ float ldsLog[TPB_TOKENS * (ROUTER_NEXP + 1)];

    const int tid  = threadIdx.x;
    const int wave = tid >> 5;
    const int lane = tid & 31;
    const int h    = lane >> 4;    // 16-lane half: selects K sub-range
    const int m    = lane & 15;    // A row / B column / C column
    const int tok0 = blockIdx.x * TPB_TOKENS + wave * 16;

    const float* xrow = x + (size_t)(tok0 + m) * ROUTER_DIM;

    // 4 accumulators: experts [0,16) [16,32) [32,48) [48,64); 16x16 f32 each
    v8f acc[4] = {};

    // prologue: stage slice 0 into buffer 0
    stage_w_slice(Wbf, &ldsW[0][0], 0, tid);
    wait_async_lds();
    __syncthreads();

    int buf = 0;
    for (int kb = 0; kb < ROUTER_DIM; kb += 32) {
        // kick off async staging of the NEXT slice into the other buffer
        if (kb + 32 < ROUTER_DIM)
            stage_w_slice(Wbf, &ldsW[buf ^ 1][0], kb + 32, tid);

        // --- A fragment: 16-bit A 16x32 layout -----------------------------
        // lanes 0-15 (h=0): K=0..15 contiguous (matches B)
        // lanes 16-31(h=1): K=16..31 contiguous
        v16bf a;
        {
            const float4 f0 = *reinterpret_cast<const float4*>(xrow + kb + h * 16);
            const float4 f1 = *reinterpret_cast<const float4*>(xrow + kb + h * 16 + 4);
            const float4 f2 = *reinterpret_cast<const float4*>(xrow + kb + h * 16 + 8);
            const float4 f3 = *reinterpret_cast<const float4*>(xrow + kb + h * 16 + 12);
            a[0]  = (__bf16)f0.x; a[1]  = (__bf16)f0.y; a[2]  = (__bf16)f0.z; a[3]  = (__bf16)f0.w;
            a[4]  = (__bf16)f1.x; a[5]  = (__bf16)f1.y; a[6]  = (__bf16)f1.z; a[7]  = (__bf16)f1.w;
            a[8]  = (__bf16)f2.x; a[9]  = (__bf16)f2.y; a[10] = (__bf16)f2.z; a[11] = (__bf16)f2.w;
            a[12] = (__bf16)f3.x; a[13] = (__bf16)f3.y; a[14] = (__bf16)f3.z; a[15] = (__bf16)f3.w;
        }
        __builtin_prefetch(xrow + kb + 32, 0, 0);   // next slice of x row

        // --- B fragments: load all four, then issue WMMAs back-to-back -----
        // B 32x16, contiguous k order chosen to match A: lanes 0-15 hold K=0..15, lanes 16-31 K=16..31;
        // VGPR j packs K=2j,2j+1 -> 16 contiguous bf16 per lane in LDS.
        const __bf16* wsl = &ldsW[buf][0];
        const v16bf b0 = *reinterpret_cast<const v16bf*>(&wsl[( 0 + m) * 32 + h * 16]);
        const v16bf b1 = *reinterpret_cast<const v16bf*>(&wsl[(16 + m) * 32 + h * 16]);
        const v16bf b2 = *reinterpret_cast<const v16bf*>(&wsl[(32 + m) * 32 + h * 16]);
        const v16bf b3 = *reinterpret_cast<const v16bf*>(&wsl[(48 + m) * 32 + h * 16]);

        acc[0] = __builtin_amdgcn_wmma_f32_16x16x32_bf16(false, a, false, b0, (short)0, acc[0], false, false);
        acc[1] = __builtin_amdgcn_wmma_f32_16x16x32_bf16(false, a, false, b1, (short)0, acc[1], false, false);
        acc[2] = __builtin_amdgcn_wmma_f32_16x16x32_bf16(false, a, false, b2, (short)0, acc[2], false, false);
        acc[3] = __builtin_amdgcn_wmma_f32_16x16x32_bf16(false, a, false, b3, (short)0, acc[3], false, false);

        // drain async staging, make next buffer visible block-wide
        wait_async_lds();
        __syncthreads();
        buf ^= 1;
    }

    // --- spill logits tile to LDS (C layout: lane m = col, row = r + 8*h) --
#pragma unroll
    for (int g = 0; g < 4; ++g) {
#pragma unroll
        for (int r = 0; r < 8; ++r) {
            ldsLog[(wave * 16 + r + 8 * h) * (ROUTER_NEXP + 1) + g * 16 + m] = acc[g][r];
        }
    }
    __syncthreads();

    // --- fused softmax + top-2 + histogram: one thread per token -----------
    if (tid < TPB_TOKENS) {
        const float* row = &ldsLog[tid * (ROUTER_NEXP + 1)];
        float maxv = row[0];
#pragma unroll 8
        for (int e = 1; e < ROUTER_NEXP; ++e) maxv = fmaxf(maxv, row[e]);

        float sum = 0.0f;
        float b1 = -INFINITY, b2 = -INFINITY;
        int   i1 = 0, i2 = 0;
        for (int e = 0; e < ROUTER_NEXP; ++e) {
            const float v = row[e];
            sum += __expf(v - maxv);
            if (v > b1)      { b2 = b1; i2 = i1; b1 = v; i1 = e; }
            else if (v > b2) { b2 = v; i2 = e; }
        }
        const float inv = 1.0f / sum;
        const int tg = blockIdx.x * TPB_TOKENS + tid;
        topw_out[2 * tg]     = __expf(b1 - maxv) * inv;
        topw_out[2 * tg + 1] = __expf(b2 - maxv) * inv;
        sel_flat[2 * tg]     = i1;
        sel_flat[2 * tg + 1] = i2;
        atomicAdd(&counts[i1], 1);
        atomicAdd(&counts[i2], 1);
    }
}

// ---------------------------------------------------------------------------
__global__ void router_init_counts(int* counts)
{
    counts[threadIdx.x] = 0;
}

// exclusive scan over 64 expert counts; also emit counts as float output
__global__ void router_scan_kernel(const int* __restrict__ counts,
                                   int* __restrict__ offsets,
                                   float* __restrict__ counts_out)
{
    if (threadIdx.x == 0) {
        int run = 0;
        for (int e = 0; e < ROUTER_NEXP; ++e) { offsets[e] = run; run += counts[e]; }
    }
    counts_out[threadIdx.x] = (float)counts[threadIdx.x];
}

// stable counting-sort scatter: one wave per expert, ballot prefix-popcount
__global__ __launch_bounds__(32)
void router_scatter_kernel(const int* __restrict__ sel_flat,
                           const int* __restrict__ offsets,
                           float* __restrict__ gather_out,
                           int n_sel)
{
    const int e    = blockIdx.x;
    const int lane = threadIdx.x;        // wave32
    int base = offsets[e];
    const unsigned lt_mask = (lane == 0) ? 0u : ((1u << lane) - 1u);
    for (int i0 = 0; i0 < n_sel; i0 += 32) {
        const int  i    = i0 + lane;
        const bool pred = (sel_flat[i] == e);
        const unsigned mask = __builtin_amdgcn_ballot_w32(pred);
        if (pred) {
            const int rank = __popc(mask & lt_mask);
            gather_out[base + rank] = (float)i;
        }
        base += __popc(mask);
    }
}

// ---------------------------------------------------------------------------
extern "C" void kernel_launch(void* const* d_in, const int* in_sizes, int n_in,
                              void* d_out, int out_size, void* d_ws, size_t ws_size,
                              hipStream_t stream)
{
    const float* x = (const float*)d_in[0];   // [N, 4096] fp32
    const float* W = (const float*)d_in[1];   // [64, 4096] fp32

    const int n_tokens = in_sizes[0] / ROUTER_DIM;   // 16384
    const int n_w      = in_sizes[1];                // 64*4096
    const int n_sel    = n_tokens * 2;               // N * TOPK

    // d_out (float): [top_w: n_sel][gather_indices: n_sel][counts: 64]
    float* topw_out   = (float*)d_out;
    float* gather_out = topw_out + n_sel;
    float* counts_out = gather_out + n_sel;

    // workspace: Wbf[64*4096] bf16 (512 KB), sel_flat[n_sel], counts, offsets
    __bf16* Wbf   = (__bf16*)d_ws;
    int* sel      = (int*)((char*)d_ws + (size_t)n_w * sizeof(__bf16));
    int* counts   = sel + n_sel;
    int* offsets  = counts + ROUTER_NEXP;

    router_init_counts<<<1, ROUTER_NEXP, 0, stream>>>(counts);

    router_w_to_bf16_kernel<<<n_w / (BLOCK_THREADS * 8), BLOCK_THREADS, 0, stream>>>(
        W, Wbf, n_w);

    router_fused_kernel<<<n_tokens / TPB_TOKENS, BLOCK_THREADS, 0, stream>>>(
        x, Wbf, topw_out, sel, counts);

    router_scan_kernel<<<1, ROUTER_NEXP, 0, stream>>>(counts, offsets, counts_out);

    router_scatter_kernel<<<ROUTER_NEXP, 32, 0, stream>>>(sel, offsets, gather_out, n_sel);
}
